// MSDeformCrossAttnBlock_58016418234613
// MI455X (gfx1250) — matrix operand order repaired and checked by reading, not compiled, not run on hardware
//
#include <hip/hip_runtime.h>
#include <hip/hip_bf16.h>

// ---------------------------------------------------------------------------
// MSDeformCrossAttnBlock for MI455X (gfx1250, wave32, WMMA).
// GEMMs use v_wmma_f32_16x16x32_bf16 (fp32 data converted to bf16 in-register,
// f32 accumulation). Each wave computes a 16x64 C tile: one A fragment feeds
// 4 WMMAs (4 independent accumulators), amortizing A loads + bf16 conversion.
// Weights are pre-packed into the WMMA B-fragment lane layout so the inner
// loop does one 32B contiguous load per lane per B fragment.
// ---------------------------------------------------------------------------

#define DIMC 256
#define NHEAD 8
#define NLVL 4
#define NPTS 4
#define HIDD 1024
#define HDIM 32
#define NBATCH 2
#define NQTOT 5440             // 64*64 + 32*32 + 16*16 + 8*8
#define NTOK (NBATCH * NQTOT)  // 10880  (divisible by 16)

typedef __attribute__((ext_vector_type(16))) __bf16 v16bf;
typedef __attribute__((ext_vector_type(16))) unsigned short v16u;
typedef __attribute__((ext_vector_type(8))) float v8f;

static __device__ __forceinline__ unsigned short f2bf(float f) {
  unsigned int u = __builtin_bit_cast(unsigned int, f);
  u += 0x7fffu + ((u >> 16) & 1u);  // round-to-nearest-even
  return (unsigned short)(u >> 16);
}

// ---------------------------------------------------------------------------
// [B, C, H*W] -> [B, H*W, C] flatten (query path)
// ---------------------------------------------------------------------------
__global__ void flatten_q_kernel(const float* __restrict__ ql,
                                 float* __restrict__ query, int n_l, int off) {
  const int total = NBATCH * n_l * DIMC;
  for (int i = blockIdx.x * blockDim.x + threadIdx.x; i < total;
       i += gridDim.x * blockDim.x) {
    const int c = i % DIMC;
    const int pos = (i / DIMC) % n_l;
    const int b = i / (DIMC * n_l);
    query[((size_t)b * NQTOT + off + pos) * DIMC + c] =
        ql[((size_t)b * DIMC + c) * n_l + pos];
  }
}

// ---------------------------------------------------------------------------
// value: [B, C, H*W] -> layernorm over C -> [B, H*W, C]
// one block (256 threads) per (b, pos); thread c handles channel c
// ---------------------------------------------------------------------------
__global__ __launch_bounds__(256) void ln_flatten_kernel(
    const float* __restrict__ v, const float* __restrict__ g,
    const float* __restrict__ beta, float* __restrict__ vn, int n_l, int off) {
  const int b = blockIdx.x / n_l;
  const int pos = blockIdx.x - b * n_l;
  const int c = threadIdx.x;
  const float x = v[((size_t)b * DIMC + c) * n_l + pos];

  float s = x, q2 = x * x;
  #pragma unroll
  for (int o = 16; o > 0; o >>= 1) {
    s += __shfl_down(s, o, 32);
    q2 += __shfl_down(q2, o, 32);
  }
  __shared__ float ss[8];
  __shared__ float sq[8];
  const int lane = threadIdx.x & 31, wid = threadIdx.x >> 5;
  if (lane == 0) { ss[wid] = s; sq[wid] = q2; }
  __syncthreads();
  if (threadIdx.x == 0) {
    float S = 0.f, Q = 0.f;
    #pragma unroll
    for (int i = 0; i < 8; ++i) { S += ss[i]; Q += sq[i]; }
    const float m = S / (float)DIMC;
    ss[0] = m;
    sq[0] = Q / (float)DIMC - m * m;
  }
  __syncthreads();
  const float y = (x - ss[0]) * rsqrtf(sq[0] + 1e-5f) * g[c] + beta[c];
  vn[((size_t)b * NQTOT + off + pos) * DIMC + c] = y;
}

// ---------------------------------------------------------------------------
// Pack fp32 weight [K, N] (row-major) into bf16 WMMA B-fragment layout:
//   fragment(nt16, kt) = 32 lanes x 16 bf16 (contiguous per lane).
//   B 32x16 16-bit layout: lanes 0-15 -> N = nt16*16+lane, K = kt*32 + {0..7, 16..23}
//                          lanes 16-31 -> same N,          K = kt*32 + {8..15, 24..31}
// Fragment order: frag = nt16 * (K/32) + kt.
// ---------------------------------------------------------------------------
__global__ void pack_wb_kernel(const float* __restrict__ W,
                               unsigned short* __restrict__ P, int K, int N) {
  const int total = K * N;
  for (int i = blockIdx.x * blockDim.x + threadIdx.x; i < total;
       i += gridDim.x * blockDim.x) {
    const int j = i & 15;
    const int lane = (i >> 4) & 31;
    const int frag = i >> 9;
    const int nkt = K >> 5;
    const int nt = frag / nkt;
    const int kt = frag - nt * nkt;
    const int n = nt * 16 + (lane & 15);
    const int k = kt * 32 + ((lane >> 4) << 3) + (j & 7) + ((j >> 3) << 4);
    P[i] = f2bf(W[(size_t)k * N + n]);
  }
}

// ---------------------------------------------------------------------------
// WMMA GEMM: C[M,N] = A[M,K] (fp32, row-major, bf16-converted on the fly)
//                     @ Bp (pre-packed bf16 fragments) + bias [+ residual]
// One wave computes one 16x64 C tile (4 WMMAs per k-step, shared A fragment).
// Wave-uniform control flow only (WMMA requires EXEC all-ones).
// A 16x32 16-bit fragment per ISA layout: lane row = lane&15;
//   lanes 0-15 hold K = {0..7, 16..23}, lanes 16-31 hold K = {8..15, 24..31}.
// ---------------------------------------------------------------------------
template <int KK, bool RES>
__global__ __launch_bounds__(256) void gemm_bf16_wmma_kernel(
    const float* __restrict__ A, const unsigned short* __restrict__ Bp,
    const float* __restrict__ bias, const float* __restrict__ R,
    float* __restrict__ C, int M, int N) {
  const int lane = threadIdx.x & 31;
  const int gw = (int)((blockIdx.x * blockDim.x + threadIdx.x) >> 5);
  const int ntc = N >> 6;  // 64-wide column tiles
  const int mt = gw / ntc;
  const int nt = gw - mt * ntc;
  if (mt >= (M >> 4)) return;  // wave-uniform exit

  constexpr int NKT = KK >> 5;  // k-steps
  const int row = lane & 15;
  const int kb = (lane >> 4) << 3;  // 0 or 8
  const float* arow = A + (size_t)(mt * 16 + row) * KK + kb;
  // four consecutive 16-wide B fragment columns
  const unsigned short* bbase =
      Bp + ((size_t)(nt * 4) * NKT) * 512 + lane * 16;

  v8f acc0 = {0.f, 0.f, 0.f, 0.f, 0.f, 0.f, 0.f, 0.f};
  v8f acc1 = acc0, acc2 = acc0, acc3 = acc0;

  #pragma unroll 4
  for (int kt = 0; kt < NKT; ++kt) {
    const int k0 = kt * 32;
    __builtin_prefetch((const void*)(arow + k0 + 32), 0, 1);  // global_prefetch
    const float4 a0 = *reinterpret_cast<const float4*>(arow + k0);
    const float4 a1 = *reinterpret_cast<const float4*>(arow + k0 + 4);
    const float4 a2 = *reinterpret_cast<const float4*>(arow + k0 + 16);
    const float4 a3 = *reinterpret_cast<const float4*>(arow + k0 + 20);
    v16u au;
    au[0] = f2bf(a0.x);  au[1] = f2bf(a0.y);  au[2] = f2bf(a0.z);  au[3] = f2bf(a0.w);
    au[4] = f2bf(a1.x);  au[5] = f2bf(a1.y);  au[6] = f2bf(a1.z);  au[7] = f2bf(a1.w);
    au[8] = f2bf(a2.x);  au[9] = f2bf(a2.y);  au[10] = f2bf(a2.z); au[11] = f2bf(a2.w);
    au[12] = f2bf(a3.x); au[13] = f2bf(a3.y); au[14] = f2bf(a3.z); au[15] = f2bf(a3.w);
    const v16bf af = __builtin_bit_cast(v16bf, au);

    const unsigned short* bk = bbase + (size_t)kt * 512;
    const v16u bu0 = *reinterpret_cast<const v16u*>(bk);
    const v16u bu1 = *reinterpret_cast<const v16u*>(bk + (size_t)NKT * 512);
    const v16u bu2 = *reinterpret_cast<const v16u*>(bk + (size_t)2 * NKT * 512);
    const v16u bu3 = *reinterpret_cast<const v16u*>(bk + (size_t)3 * NKT * 512);

    acc0 = __builtin_amdgcn_wmma_f32_16x16x32_bf16(
        false, af, false, __builtin_bit_cast(v16bf, bu0), (short)0, acc0,
        false, false);
    acc1 = __builtin_amdgcn_wmma_f32_16x16x32_bf16(
        false, af, false, __builtin_bit_cast(v16bf, bu1), (short)0, acc1,
        false, false);
    acc2 = __builtin_amdgcn_wmma_f32_16x16x32_bf16(
        false, af, false, __builtin_bit_cast(v16bf, bu2), (short)0, acc2,
        false, false);
    acc3 = __builtin_amdgcn_wmma_f32_16x16x32_bf16(
        false, af, false, __builtin_bit_cast(v16bf, bu3), (short)0, acc3,
        false, false);
  }

  const int r0 = mt * 16 + ((lane >> 4) << 3);
  const int cbase = nt * 64 + (lane & 15);
  auto epi = [&](const v8f& a, int c) {
    const float bv = bias[c];
    #pragma unroll
    for (int r = 0; r < 8; ++r) {
      const size_t idx = (size_t)(r0 + r) * N + c;
      float vv = a[r] + bv;
      if constexpr (RES) vv += R[idx];
      C[idx] = vv;
    }
  };
  epi(acc0, cbase);
  epi(acc1, cbase + 16);
  epi(acc2, cbase + 32);
  epi(acc3, cbase + 48);
}

// ---------------------------------------------------------------------------
// Deformable attention sampling.
// One wave per (b, q, head); lane = head channel (HD=32 -> coalesced 128B
// gathers of val). Softmax over the 16 (level,point) logits in-register.
// val layout: [B*Nv, NH*HD] row-major; out: [B*Nq, 256].
// ---------------------------------------------------------------------------
__global__ __launch_bounds__(256) void ms_deform_sample_kernel(
    const float* __restrict__ val, const float* __restrict__ offr,
    const float* __restrict__ awr, float* __restrict__ out) {
  const int lane = threadIdx.x & 31;
  const int gw = (int)((blockIdx.x * blockDim.x + threadIdx.x) >> 5);
  if (gw >= NBATCH * NQTOT * NHEAD) return;
  const int h = gw & (NHEAD - 1);
  const int q = (gw >> 3) % NQTOT;
  const int b = gw / (NQTOT * NHEAD);

  // query's own level geometry -> reference point
  int base, wl;
  if (q < 4096)      { base = 0;    wl = 64; }
  else if (q < 5120) { base = 4096; wl = 32; }
  else if (q < 5376) { base = 5120; wl = 16; }
  else               { base = 5376; wl = 8;  }
  const int pos = q - base;
  const float rx = ((float)(pos % wl) + 0.5f) / (float)wl;
  const float ry = ((float)(pos / wl) + 0.5f) / (float)wl;  // square levels

  // softmax over NL*NP = 16 attention logits
  const float* awp = awr + (size_t)(b * NQTOT + q) * (NHEAD * NLVL * NPTS) +
                     h * (NLVL * NPTS);
  float e[16];
  float mx = awp[0];
  #pragma unroll
  for (int i = 1; i < 16; ++i) mx = fmaxf(mx, awp[i]);
  float ssum = 0.f;
  #pragma unroll
  for (int i = 0; i < 16; ++i) { e[i] = __expf(awp[i] - mx); ssum += e[i]; }
  const float inv = 1.f / ssum;

  const float* offp = offr + (size_t)(b * NQTOT + q) * DIMC +
                      h * (NLVL * NPTS * 2);

  const int lbase[4] = {0, 4096, 5120, 5376};
  const int lw[4] = {64, 32, 16, 8};
  float accv = 0.f;
  #pragma unroll
  for (int l = 0; l < NLVL; ++l) {
    const int wlv = lw[l];
    const float* vbase =
        val + (((size_t)b * NQTOT + lbase[l]) * NHEAD + h) * HDIM + lane;
    auto fetch = [&](int xi, int yi) -> float {
      if (xi < 0 || yi < 0 || xi >= wlv || yi >= wlv) return 0.f;
      return vbase[(size_t)(yi * wlv + xi) * (NHEAD * HDIM)];
    };
    #pragma unroll
    for (int p = 0; p < NPTS; ++p) {
      // loc*W - 0.5 == ref*W + off - 0.5  (since off is divided by (W,H))
      const float sx = rx * (float)wlv + offp[(l * 4 + p) * 2 + 0] - 0.5f;
      const float sy = ry * (float)wlv + offp[(l * 4 + p) * 2 + 1] - 0.5f;
      const float fx = floorf(sx), fy = floorf(sy);
      const int x0 = (int)fx, y0 = (int)fy;
      const float lx = sx - fx, ly = sy - fy;
      const float g00 = fetch(x0, y0);
      const float g10 = fetch(x0 + 1, y0);
      const float g01 = fetch(x0, y0 + 1);
      const float g11 = fetch(x0 + 1, y0 + 1);
      const float bil = (1.f - lx) * (1.f - ly) * g00 + lx * (1.f - ly) * g10 +
                        (1.f - lx) * ly * g01 + lx * ly * g11;
      accv += (e[l * 4 + p] * inv) * bil;
    }
  }
  out[(size_t)(b * NQTOT + q) * DIMC + h * HDIM + lane] = accv;
}

// ---------------------------------------------------------------------------
// Per-level 3x3 depthwise conv (SAME, zero pad) + bias + exact GELU.
// h1 layout: [B*Nq, HID]
// ---------------------------------------------------------------------------
__global__ void dwconv_gelu_kernel(const float* __restrict__ h1,
                                   const float* __restrict__ Wdw,
                                   const float* __restrict__ bdw,
                                   float* __restrict__ outp, int n_l, int off,
                                   int wl) {
  const int total = NBATCH * n_l * HIDD;
  for (int i = blockIdx.x * blockDim.x + threadIdx.x; i < total;
       i += gridDim.x * blockDim.x) {
    const int c = i % HIDD;
    const int pos = (i / HIDD) % n_l;
    const int b = i / (HIDD * n_l);
    const int py = pos / wl, px = pos - (pos / wl) * wl;
    float acc = bdw[c];
    #pragma unroll
    for (int dy = -1; dy <= 1; ++dy) {
      #pragma unroll
      for (int dx = -1; dx <= 1; ++dx) {
        const int yy = py + dy, xx = px + dx;
        if (yy >= 0 && yy < wl && xx >= 0 && xx < wl)
          acc += h1[((size_t)b * NQTOT + off + yy * wl + xx) * HIDD + c] *
                 Wdw[((dy + 1) * 3 + (dx + 1)) * HIDD + c];
      }
    }
    const float gel = 0.5f * acc * (1.f + erff(acc * 0.70710678118654752f));
    outp[((size_t)b * NQTOT + off + pos) * HIDD + c] = gel;
  }
}

// ---------------------------------------------------------------------------
// Host launch
// ---------------------------------------------------------------------------
extern "C" void kernel_launch(void* const* d_in, const int* in_sizes, int n_in,
                              void* d_out, int out_size, void* d_ws,
                              size_t ws_size, hipStream_t stream) {
  (void)in_sizes; (void)n_in; (void)out_size; (void)ws_size;

  // setup_inputs() dict order: q0,v0,q1,v1,q2,v2,q3,v3, vn_g, vn_b, ...
  const float* q[4] = {(const float*)d_in[0], (const float*)d_in[2],
                       (const float*)d_in[4], (const float*)d_in[6]};
  const float* v[4] = {(const float*)d_in[1], (const float*)d_in[3],
                       (const float*)d_in[5], (const float*)d_in[7]};
  const float* vn_g = (const float*)d_in[8];
  const float* vn_b = (const float*)d_in[9];
  const float* W_off = (const float*)d_in[10];
  const float* b_off = (const float*)d_in[11];
  const float* W_attn = (const float*)d_in[12];
  const float* b_attn = (const float*)d_in[13];
  const float* W_val = (const float*)d_in[14];
  const float* b_val = (const float*)d_in[15];
  const float* W_out = (const float*)d_in[16];
  const float* b_out = (const float*)d_in[17];
  const float* W_fc1 = (const float*)d_in[18];
  const float* b_fc1 = (const float*)d_in[19];
  const float* W_dw = (const float*)d_in[20];
  const float* b_dw = (const float*)d_in[21];
  const float* W_fc2 = (const float*)d_in[22];
  const float* b_fc2 = (const float*)d_in[23];

  // workspace bump allocator
  char* ws = (char*)d_ws;
  size_t cur = 0;
  auto walloc = [&](size_t bytes) -> void* {
    void* p = ws + cur;
    cur += (bytes + 255) & ~(size_t)255;
    return p;
  };
  float* query = (float*)walloc((size_t)NTOK * DIMC * 4);
  float* vn = (float*)walloc((size_t)NTOK * DIMC * 4);
  float* val = (float*)walloc((size_t)NTOK * DIMC * 4);
  float* offb = (float*)walloc((size_t)NTOK * DIMC * 4);
  float* awb = (float*)walloc((size_t)NTOK * (NHEAD * NLVL * NPTS) * 4);
  float* xb = (float*)walloc((size_t)NTOK * DIMC * 4);
  float* h1 = (float*)walloc((size_t)NTOK * HIDD * 4);
  float* h1c = (float*)walloc((size_t)NTOK * HIDD * 4);
  unsigned short* Pval = (unsigned short*)walloc((size_t)DIMC * DIMC * 2);
  unsigned short* Poff = (unsigned short*)walloc((size_t)DIMC * DIMC * 2);
  unsigned short* Pattn = (unsigned short*)walloc((size_t)DIMC * 128 * 2);
  unsigned short* Pout = (unsigned short*)walloc((size_t)DIMC * DIMC * 2);
  unsigned short* Pfc1 = (unsigned short*)walloc((size_t)DIMC * HIDD * 2);
  unsigned short* Pfc2 = (unsigned short*)walloc((size_t)HIDD * DIMC * 2);
  float* attn = vn;  // vn is dead after the W_val GEMM -> reuse for attn_src

  const int wls[4] = {64, 32, 16, 8};
  const int nls[4] = {4096, 1024, 256, 64};
  const int offs[4] = {0, 4096, 5120, 5376};

  // 1) flatten q, layernorm+flatten v
  for (int l = 0; l < 4; ++l) {
    int total = NBATCH * nls[l] * DIMC;
    flatten_q_kernel<<<(total + 255) / 256, 256, 0, stream>>>(q[l], query,
                                                              nls[l], offs[l]);
    ln_flatten_kernel<<<NBATCH * nls[l], 256, 0, stream>>>(
        v[l], vn_g, vn_b, vn, nls[l], offs[l]);
  }

  // 2) pack weights into bf16 WMMA fragments
  auto pack = [&](const float* W, unsigned short* P, int K, int N) {
    int total = K * N;
    pack_wb_kernel<<<(total + 255) / 256, 256, 0, stream>>>(W, P, K, N);
  };
  pack(W_val, Pval, DIMC, DIMC);
  pack(W_off, Poff, DIMC, DIMC);
  pack(W_attn, Pattn, DIMC, 128);
  pack(W_out, Pout, DIMC, DIMC);
  pack(W_fc1, Pfc1, DIMC, HIDD);
  pack(W_fc2, Pfc2, HIDD, DIMC);

  // 3) GEMM helper: one wave per 16x64 tile, 8 waves per block
  auto gemm = [&](const float* A, const unsigned short* P, const float* bias,
                  const float* R, float* C, int M, int N, int K) {
    int waves = (M >> 4) * (N >> 6);
    int blocks = (waves + 7) / 8;
    if (K == 1024) {
      if (R)
        gemm_bf16_wmma_kernel<1024, true>
            <<<blocks, 256, 0, stream>>>(A, P, bias, R, C, M, N);
      else
        gemm_bf16_wmma_kernel<1024, false>
            <<<blocks, 256, 0, stream>>>(A, P, bias, nullptr, C, M, N);
    } else {
      if (R)
        gemm_bf16_wmma_kernel<256, true>
            <<<blocks, 256, 0, stream>>>(A, P, bias, R, C, M, N);
      else
        gemm_bf16_wmma_kernel<256, false>
            <<<blocks, 256, 0, stream>>>(A, P, bias, nullptr, C, M, N);
    }
  };

  gemm(vn, Pval, b_val, nullptr, val, NTOK, DIMC, DIMC);      // value proj
  gemm(query, Poff, b_off, nullptr, offb, NTOK, DIMC, DIMC);  // offsets
  gemm(query, Pattn, b_attn, nullptr, awb, NTOK, 128, DIMC);  // attn logits

  // 4) deformable sampling -> attn_src [B*Nq, 256]
  {
    int waves = NBATCH * NQTOT * NHEAD;
    ms_deform_sample_kernel<<<(waves + 7) / 8, 256, 0, stream>>>(val, offb,
                                                                 awb, attn);
  }

  // 5) output proj + residual(query) -> x
  gemm(attn, Pout, b_out, query, xb, NTOK, DIMC, DIMC);

  // 6) MixFFN: fc1 -> per-level dwconv3x3 + GELU -> fc2 + residual(x)
  gemm(xb, Pfc1, b_fc1, nullptr, h1, NTOK, HIDD, DIMC);
  for (int l = 0; l < 4; ++l) {
    int total = NBATCH * nls[l] * HIDD;
    dwconv_gelu_kernel<<<(total + 255) / 256, 256, 0, stream>>>(
        h1, W_dw, b_dw, h1c, nls[l], offs[l], wls[l]);
  }
  gemm(h1c, Pfc2, b_fc2, xb, (float*)d_out, NTOK, DIMC, HIDD);
}
